// BiLSTMCRF_16879221473617
// MI455X (gfx1250) — matrix operand order repaired and checked by reading, not compiled, not run on hardware
//
#include <hip/hip_runtime.h>
#include <cstdint>
#include <cstddef>

// ------------------------- problem dimensions -------------------------
constexpr int kB  = 128;           // batch
constexpr int kS  = 128;           // sequence length
constexpr int kWC = 20;            // chars per word
constexpr int kT  = 9;             // tags
constexpr int kDE = 300, kDC = 30, kHC = 25, kH = 256;
constexpr int kBS = kB * kS;       // 16384 rows
constexpr int kD0 = kDE + kHC;     // 325
constexpr int kD0P = 352;          // padded to multiple of 32 for WMMA K
constexpr int kD1 = 2 * kH;        // 512
constexpr int kG  = 4 * kH;        // 1024 gate outputs

// ------------------------- WMMA vector types --------------------------
typedef __attribute__((ext_vector_type(16))) __bf16 v16bf;
typedef __attribute__((ext_vector_type(8)))  __bf16 v8bf;
typedef __attribute__((ext_vector_type(8)))  float  v8f;

__device__ __forceinline__ unsigned short f2bf(float f) {
  unsigned int u = __float_as_uint(f);
  u += 0x7fffu + ((u >> 16) & 1u);        // round-to-nearest-even
  return (unsigned short)(u >> 16);
}
__device__ __forceinline__ float sigm(float x) { return 1.f / (1.f + __expf(-x)); }

__device__ __forceinline__ v16bf mk_a(const unsigned short* p) {
  v8bf lo = *(const v8bf*)(p);
  v8bf hi = *(const v8bf*)(p + 16);
  v16bf a;
#pragma unroll
  for (int i = 0; i < 8; ++i) { a[i] = lo[i]; a[8 + i] = hi[i]; }
  return a;
}
__device__ __forceinline__ v16bf mk_b(const unsigned short* p, int ld) {
  v16bf b;
#pragma unroll
  for (int i = 0; i < 8; ++i) {
    unsigned short u0 = p[(size_t)i * ld], u1 = p[(size_t)(16 + i) * ld];
    b[i] = __builtin_bit_cast(__bf16, u0); b[8 + i] = __builtin_bit_cast(__bf16, u1);
  }
  return b;
}
// ======================================================================
// Generic bf16 GEMM:  C[M,N](f32) = A[M,K](bf16) * Bt[K,N](bf16, weight
// pre-transposed to [K,N]).  One wave computes a 16x64 tile (one A
// fragment feeds 4 WMMAs); 4 waves/block stack along M (64 rows).
// grid = (N/64, M/64).  K mult of 32, N mult of 64, M mult of 64.
// ======================================================================
__global__ __launch_bounds__(128) void gemm_bf16(
    const unsigned short* __restrict__ A,
    const unsigned short* __restrict__ Bt,
    float* __restrict__ C, int K, int lda, int N) {
  const int lane  = threadIdx.x & 31;
  const int wave  = threadIdx.x >> 5;
  const int mbase = blockIdx.y * 64 + wave * 16;
  const int nbase = blockIdx.x * 64;
  const int ahalf = (lane & 16) ? 8 : 0;       // 16-bit A VGPR layout halves
  const unsigned short* Ap = A + (size_t)(mbase + (lane & 15)) * lda + ahalf;
  const unsigned short* Bp = Bt + (size_t)ahalf * N + nbase + (lane & 15);  // lane indexes N
  const size_t bstep = (size_t)32 * N;
  v8f acc0 = {}, acc1 = {}, acc2 = {}, acc3 = {};
  for (int k = 0; k < K; k += 32) {
    v16bf a  = mk_a(Ap);
    v16bf b0 = mk_b(Bp, N);
    v16bf b1 = mk_b(Bp + 16, N);
    v16bf b2 = mk_b(Bp + 32, N);
    v16bf b3 = mk_b(Bp + 48, N);
    __builtin_prefetch(Bp + bstep, 0, 0);
    acc0 = __builtin_amdgcn_wmma_f32_16x16x32_bf16(false, a, false, b0, (short)0, acc0, false, false);
    acc1 = __builtin_amdgcn_wmma_f32_16x16x32_bf16(false, a, false, b1, (short)0, acc1, false, false);
    acc2 = __builtin_amdgcn_wmma_f32_16x16x32_bf16(false, a, false, b2, (short)0, acc2, false, false);
    acc3 = __builtin_amdgcn_wmma_f32_16x16x32_bf16(false, a, false, b3, (short)0, acc3, false, false);
    Ap += 32;
    Bp += bstep;
  }
  float* Cp = C + (size_t)(mbase + ((lane & 16) ? 8 : 0)) * N + nbase + (lane & 15);
#pragma unroll
  for (int r = 0; r < 8; ++r) {
    Cp[0] = acc0[r]; Cp[16] = acc1[r]; Cp[32] = acc2[r]; Cp[48] = acc3[r];
    Cp += N;
  }
}

// ======================================================================
// Fused recurrent step: for one timestep, compute h@Whh^T for both
// directions (rows 0..127 fwd, 128..255 bwd) AND apply the LSTM gates.
// A wave's 4 accumulator tiles sit at column offsets {0,H,2H,3H}, i.e. the
// i,f,g,o gates of the same 16x16 block of hidden units, so the epilogue
// can finish c/h locally.  Hidden state double-buffered (hin -> hnew) to
// avoid intra-launch read/write races.  grid = (kH/16, 256/64), block 128.
// ======================================================================
__global__ __launch_bounds__(128) void fused_step(
    const unsigned short* __restrict__ hin,   // [256, kH] bf16
    const unsigned short* __restrict__ BtF,   // Whh fwd, [kH, kG]
    const unsigned short* __restrict__ BtB,   // Whh bwd, [kH, kG]
    const float* __restrict__ preF, const float* __restrict__ preB,
    const float* __restrict__ bF, const float* __restrict__ bB,
    float* __restrict__ c_buf,                 // [256, kH]
    unsigned short* __restrict__ hnew,         // [256, kH] bf16
    float* __restrict__ houtF, float* __restrict__ houtBrev, int t) {
  const int lane  = threadIdx.x & 31;
  const int wave  = threadIdx.x >> 5;
  const int mbase = blockIdx.y * 64 + wave * 16;
  const int nbase = blockIdx.x * 16;           // hidden-unit tile
  const int dsel  = (mbase >= kB);
  const unsigned short* Bt = dsel ? BtB : BtF;
  const int ahalf = (lane & 16) ? 8 : 0;
  const unsigned short* Ap = hin + (size_t)(mbase + (lane & 15)) * kH + ahalf;
  const unsigned short* Bp = Bt + (size_t)ahalf * kG + nbase + (lane & 15);
  const size_t bstep = (size_t)32 * kG;
  v8f acc0 = {}, acc1 = {}, acc2 = {}, acc3 = {};
#pragma unroll 2
  for (int k = 0; k < kH; k += 32) {
    v16bf a  = mk_a(Ap);
    v16bf b0 = mk_b(Bp, kG);                 // gate i columns
    v16bf b1 = mk_b(Bp + kH, kG);            // gate f
    v16bf b2 = mk_b(Bp + 2 * kH, kG);        // gate g
    v16bf b3 = mk_b(Bp + 3 * kH, kG);        // gate o
    acc0 = __builtin_amdgcn_wmma_f32_16x16x32_bf16(false, a, false, b0, (short)0, acc0, false, false);
    acc1 = __builtin_amdgcn_wmma_f32_16x16x32_bf16(false, a, false, b1, (short)0, acc1, false, false);
    acc2 = __builtin_amdgcn_wmma_f32_16x16x32_bf16(false, a, false, b2, (short)0, acc2, false, false);
    acc3 = __builtin_amdgcn_wmma_f32_16x16x32_bf16(false, a, false, b3, (short)0, acc3, false, false);
    Ap += 32;
    Bp += bstep;
  }
  // ---- gate epilogue ----
  const int colj = nbase + (lane & 15);
  const int row0 = mbase + ((lane & 16) ? 8 : 0);
  const float* pre  = dsel ? preB : preF;
  const float* bias = dsel ? bB : bF;
  float* hout = dsel ? houtBrev : houtF;
  const float bi = bias[colj], bf = bias[kH + colj],
              bg = bias[2 * kH + colj], bo = bias[3 * kH + colj];
#pragma unroll
  for (int r = 0; r < 8; ++r) {
    const int row = row0 + r;            // 0..255
    const int b   = row & (kB - 1);
    const size_t prow = (size_t)(b * kS + t) * kG;
    float zi = acc0[r] + pre[prow + colj]          + bi;
    float zf = acc1[r] + pre[prow + kH + colj]     + bf;
    float zg = acc2[r] + pre[prow + 2 * kH + colj] + bg;
    float zo = acc3[r] + pre[prow + 3 * kH + colj] + bo;
    float c = sigm(zf) * c_buf[(size_t)row * kH + colj] + sigm(zi) * tanhf(zg);
    float h = sigm(zo) * tanhf(c);
    c_buf[(size_t)row * kH + colj] = c;
    hnew[(size_t)row * kH + colj] = f2bf(h);
    hout[(size_t)(b * kS + t) * kH + colj] = h;
  }
}

// ======================================================================
// Weight repack: f32 [N,K] row-major -> bf16 [KP,N] (transposed, zero pad)
// ======================================================================
__global__ void packT(const float* __restrict__ in, unsigned short* __restrict__ out,
                      int N, int K, int KP) {
  int tid = blockIdx.x * blockDim.x + threadIdx.x;
  if (tid >= KP * N) return;
  int k = tid / N, n = tid - k * N;
  out[tid] = (k < K) ? f2bf(in[(size_t)n * K + k]) : (unsigned short)0;
}

// per-sequence valid lengths from the boolean mask (jnp bool_ = 1 byte)
__global__ void lengths_k(const unsigned char* __restrict__ mask, int* __restrict__ lens) {
  int b = threadIdx.x;
  int s = 0;
  for (int t = 0; t < kS; ++t) s += mask[b * kS + t] ? 1 : 0;
  lens[b] = s;
}

// word embedding gather -> bf16 x buffer (cols [0,300)), zero pad cols [325,352)
__global__ void embed_word(const int* __restrict__ wid, const float* __restrict__ Ww,
                           unsigned short* __restrict__ xb) {
  int tid = blockIdx.x * blockDim.x + threadIdx.x;
  if (tid >= kBS * kD0P) return;
  int row = tid / kD0P, c = tid - row * kD0P;
  if (c < kDE)        xb[tid] = f2bf(Ww[(size_t)wid[row] * kDE + c]);
  else if (c >= kD0)  xb[tid] = 0;     // pad columns; [300,325) written by char LSTM
}

// ======================================================================
// Char LSTM: one wave per word, 20 steps, D=30 H=25.  Writes last valid
// hidden state (bf16) into x columns [300,325).
// ======================================================================
__global__ __launch_bounds__(32) void charlstm(
    const int* __restrict__ cid, const float* __restrict__ Wc,
    const float* __restrict__ Wih, const float* __restrict__ Whh,
    const float* __restrict__ bias, unsigned short* __restrict__ xb) {
  const int word = blockIdx.x, lane = threadIdx.x;
  __shared__ float sx[kDC];
  __shared__ float sh[kHC];
  int myc = (lane < kWC) ? cid[word * kWC + lane] : 0;
  unsigned long long bal = __ballot(myc != 0);
  int clen = (int)__popcll(bal);
  if (clen < 1) clen = 1;
  if (lane < kHC) sh[lane] = 0.f;
  float c = 0.f;
  __syncthreads();
  for (int t = 0; t < kWC; ++t) {
    int ct = cid[word * kWC + t];
    if (lane < kDC) sx[lane] = Wc[(size_t)ct * kDC + lane];
    __syncthreads();
    float z[4];
    if (lane < kHC) {
#pragma unroll
      for (int q = 0; q < 4; ++q) {
        int g = q * kHC + lane;
        float s = bias[g];
        for (int d = 0; d < kDC; ++d) s += Wih[g * kDC + d] * sx[d];
        for (int m = 0; m < kHC; ++m) s += Whh[g * kHC + m] * sh[m];
        z[q] = s;
      }
    }
    __syncthreads();                       // all lanes done reading sh
    if (lane < kHC) {
      c = sigm(z[1]) * c + sigm(z[0]) * tanhf(z[2]);
      float h = sigm(z[3]) * tanhf(c);
      sh[lane] = h;
      if (t == clen - 1)
        xb[(size_t)word * kD0P + kDE + lane] = f2bf(h);
    }
    __syncthreads();
  }
}

// length-aware per-sequence reversal of bf16 [BS, cols]
__global__ void rev_bf16(const unsigned short* __restrict__ in,
                         unsigned short* __restrict__ out,
                         const int* __restrict__ lens, int cols) {
  int tid = blockIdx.x * blockDim.x + threadIdx.x;
  if (tid >= kBS * cols) return;
  int row = tid / cols, c = tid - row * cols;
  int b = row / kS, t = row - b * kS;
  int len = lens[b];
  int rt = (t < len) ? (len - 1 - t) : t;
  out[tid] = in[(size_t)(b * kS + rt) * cols + c];
}

// zero hidden/cell state for both directions
__global__ void zero_state(unsigned short* __restrict__ h, float* __restrict__ c) {
  int tid = blockIdx.x * blockDim.x + threadIdx.x;
  if (tid >= 2 * kB * kH) return;
  h[tid] = 0; c[tid] = 0.f;
}

// Layer-0 combine: h0cat = [fwd | unrev(bwd)] (bf16) and its per-sequence
// reversal h0rcat (input for layer-1 backward LSTM).
__global__ void combine0(const float* __restrict__ hF, const float* __restrict__ hBrev,
                         const int* __restrict__ lens,
                         unsigned short* __restrict__ hcat,
                         unsigned short* __restrict__ hrcat) {
  int tid = blockIdx.x * blockDim.x + threadIdx.x;
  if (tid >= kBS * kD1) return;
  int row = tid / kD1, c = tid - row * kD1;
  int b = row / kS, t = row - b * kS;
  int len = lens[b];
  int rt = (t < len) ? (len - 1 - t) : t;
  float v0 = (c < kH) ? hF[(size_t)row * kH + c]
                      : hBrev[(size_t)(b * kS + rt) * kH + (c - kH)];
  hcat[tid] = f2bf(v0);
  float v1 = (c < kH) ? hF[(size_t)(b * kS + rt) * kH + c]
                      : hBrev[(size_t)row * kH + (c - kH)];   // rev(rev(t)) == t
  hrcat[tid] = f2bf(v1);
}

__global__ void combine1(const float* __restrict__ hF, const float* __restrict__ hBrev,
                         const int* __restrict__ lens, unsigned short* __restrict__ hcat) {
  int tid = blockIdx.x * blockDim.x + threadIdx.x;
  if (tid >= kBS * kD1) return;
  int row = tid / kD1, c = tid - row * kD1;
  int b = row / kS, t = row - b * kS;
  int len = lens[b];
  int rt = (t < len) ? (len - 1 - t) : t;
  float v = (c < kH) ? hF[(size_t)row * kH + c]
                     : hBrev[(size_t)(b * kS + rt) * kH + (c - kH)];
  hcat[tid] = f2bf(v);
}

// tanh(ffpre + ffb) then project to 9 emissions per row (one block/row)
__global__ __launch_bounds__(512) void em_k(const float* __restrict__ ffpre,
                                            const float* __restrict__ ffb,
                                            const float* __restrict__ outW,
                                            const float* __restrict__ outb,
                                            float* __restrict__ em) {
  __shared__ float sh[kD1];
  int row = blockIdx.x, tid = threadIdx.x;
  sh[tid] = tanhf(ffpre[(size_t)row * kD1 + tid] + ffb[tid]);
  __syncthreads();
  if (tid < kT) {
    float s = outb[tid];
    for (int j = 0; j < kD1; ++j) s += sh[j] * outW[tid * kD1 + j];
    em[row * kT + tid] = s;
  }
}

// ======================================================================
// CRF NLL: one lane per sequence; numerator path score + forward algorithm
// (9-tag logsumexp); LDS tree-reduce the mean into d_out[0].
// ======================================================================
__global__ __launch_bounds__(128) void crf_k(
    const float* __restrict__ em, const int* __restrict__ tags,
    const unsigned char* __restrict__ mask, const int* __restrict__ lens,
    const float* __restrict__ startv, const float* __restrict__ endv,
    const float* __restrict__ trans, float* __restrict__ out) {
  __shared__ float red[kB];
  __shared__ float str[kT * kT];
  int b = threadIdx.x;
  if (b < kT * kT) str[b] = trans[b];
  __syncthreads();
  int len = lens[b];
  const int* tg = tags + b * kS;
  const unsigned char* mk = mask + b * kS;
  const float* eb = em + (size_t)b * kS * kT;
  // numerator (gold path score)
  float score = startv[tg[0]] + eb[tg[0]];
  int tp = tg[0];
  for (int t = 1; t < kS; ++t) {
    int tc = tg[t];
    float m = mk[t] ? 1.f : 0.f;
    score += m * (str[tp * kT + tc] + eb[t * kT + tc]);
    tp = tc;
  }
  score += endv[tg[len - 1]];
  // denominator (forward algorithm)
  float alpha[kT];
#pragma unroll
  for (int j = 0; j < kT; ++j) alpha[j] = startv[j] + eb[j];
  for (int t = 1; t < kS; ++t) {
    if (mk[t]) {
      float nxt[kT];
#pragma unroll
      for (int j = 0; j < kT; ++j) {
        float mx = -1e30f;
#pragma unroll
        for (int i = 0; i < kT; ++i) mx = fmaxf(mx, alpha[i] + str[i * kT + j]);
        float s = 0.f;
#pragma unroll
        for (int i = 0; i < kT; ++i) s += __expf(alpha[i] + str[i * kT + j] - mx);
        nxt[j] = mx + __logf(s) + eb[t * kT + j];
      }
#pragma unroll
      for (int j = 0; j < kT; ++j) alpha[j] = nxt[j];
    }
  }
  float mx = -1e30f;
#pragma unroll
  for (int j = 0; j < kT; ++j) mx = fmaxf(mx, alpha[j] + endv[j]);
  float s = 0.f;
#pragma unroll
  for (int j = 0; j < kT; ++j) s += __expf(alpha[j] + endv[j] - mx);
  float logZ = mx + __logf(s);
  red[b] = logZ - score;
  __syncthreads();
  for (int k = kB / 2; k > 0; k >>= 1) {
    if (b < k) red[b] += red[b + k];
    __syncthreads();
  }
  if (b == 0) out[0] = red[0] / (float)kB;
}

// ======================================================================
// Host-side orchestration
// ======================================================================
extern "C" void kernel_launch(void* const* d_in, const int* in_sizes, int n_in,
                              void* d_out, int out_size, void* d_ws, size_t ws_size,
                              hipStream_t stream) {
  (void)in_sizes; (void)n_in; (void)out_size; (void)ws_size;
  const int*   word_ids = (const int*)d_in[0];
  const int*   char_ids = (const int*)d_in[1];
  const int*   tags     = (const int*)d_in[2];
  const unsigned char* mask = (const unsigned char*)d_in[3];  // jnp bool_ (1B)
  const float* W_word = (const float*)d_in[4];
  const float* W_char = (const float*)d_in[5];
  const float* c_Wih = (const float*)d_in[6];
  const float* c_Whh = (const float*)d_in[7];
  const float* c_b   = (const float*)d_in[8];
  const float* l0f_Wih=(const float*)d_in[9],  *l0f_Whh=(const float*)d_in[10], *l0f_b=(const float*)d_in[11];
  const float* l0b_Wih=(const float*)d_in[12], *l0b_Whh=(const float*)d_in[13], *l0b_b=(const float*)d_in[14];
  const float* l1f_Wih=(const float*)d_in[15], *l1f_Whh=(const float*)d_in[16], *l1f_b=(const float*)d_in[17];
  const float* l1b_Wih=(const float*)d_in[18], *l1b_Whh=(const float*)d_in[19], *l1b_b=(const float*)d_in[20];
  const float* ff_W=(const float*)d_in[21], *ff_b=(const float*)d_in[22];
  const float* out_W=(const float*)d_in[23], *out_b=(const float*)d_in[24];
  const float* crf_start=(const float*)d_in[25], *crf_end=(const float*)d_in[26], *crf_trans=(const float*)d_in[27];

  // ---- workspace carving (~280 MB) ----
  size_t off = 0;
  auto carve = [&](size_t bytes) -> void* {
    void* p = (char*)d_ws + off;
    off = (off + bytes + 255) & ~(size_t)255;
    return p;
  };
  unsigned short* xb    = (unsigned short*)carve((size_t)kBS * kD0P * 2);
  unsigned short* xrb   = (unsigned short*)carve((size_t)kBS * kD0P * 2);
  unsigned short* wT0f  = (unsigned short*)carve((size_t)kD0P * kG * 2);
  unsigned short* wT0b  = (unsigned short*)carve((size_t)kD0P * kG * 2);
  unsigned short* uT0f  = (unsigned short*)carve((size_t)kH * kG * 2);
  unsigned short* uT0b  = (unsigned short*)carve((size_t)kH * kG * 2);
  unsigned short* uT1f  = (unsigned short*)carve((size_t)kH * kG * 2);
  unsigned short* uT1b  = (unsigned short*)carve((size_t)kH * kG * 2);
  unsigned short* wT1f  = (unsigned short*)carve((size_t)kD1 * kG * 2);
  unsigned short* wT1b  = (unsigned short*)carve((size_t)kD1 * kG * 2);
  unsigned short* ffT   = (unsigned short*)carve((size_t)kD1 * kD1 * 2);
  float* pre_f  = (float*)carve((size_t)kBS * kG * 4);
  float* pre_b  = (float*)carve((size_t)kBS * kG * 4);
  float* houtF  = (float*)carve((size_t)kBS * kH * 4);
  float* houtBr = (float*)carve((size_t)kBS * kH * 4);
  unsigned short* h0cat  = (unsigned short*)carve((size_t)kBS * kD1 * 2);
  unsigned short* h0rcat = (unsigned short*)carve((size_t)kBS * kD1 * 2);
  unsigned short* h1cat  = (unsigned short*)carve((size_t)kBS * kD1 * 2);
  float* ffpre  = (float*)carve((size_t)kBS * kD1 * 4);
  float* em     = (float*)carve((size_t)kBS * kT * 4);
  unsigned short* h_st0 = (unsigned short*)carve((size_t)2 * kB * kH * 2);  // double-
  unsigned short* h_st1 = (unsigned short*)carve((size_t)2 * kB * kH * 2);  // buffered h
  float* c_buf  = (float*)carve((size_t)2 * kB * kH * 4);
  int*   lens   = (int*)carve((size_t)kB * 4);

  // ---- setup ----
  lengths_k<<<1, kB, 0, stream>>>(mask, lens);
  auto PK = [&](const float* in, unsigned short* out, int N, int K, int KP) {
    int total = KP * N;
    packT<<<(total + 255) / 256, 256, 0, stream>>>(in, out, N, K, KP);
  };
  PK(l0f_Wih, wT0f, kG, kD0, kD0P);
  PK(l0b_Wih, wT0b, kG, kD0, kD0P);
  PK(l0f_Whh, uT0f, kG, kH, kH);
  PK(l0b_Whh, uT0b, kG, kH, kH);
  PK(l1f_Wih, wT1f, kG, kD1, kD1);
  PK(l1b_Wih, wT1b, kG, kD1, kD1);
  PK(l1f_Whh, uT1f, kG, kH, kH);
  PK(l1b_Whh, uT1b, kG, kH, kH);
  PK(ff_W, ffT, kD1, kD1, kD1);

  embed_word<<<(kBS * kD0P + 255) / 256, 256, 0, stream>>>(word_ids, W_word, xb);
  charlstm<<<kBS, 32, 0, stream>>>(char_ids, W_char, c_Wih, c_Whh, c_b, xb);
  rev_bf16<<<(kBS * kD0P + 255) / 256, 256, 0, stream>>>(xb, xrb, lens, kD0P);

  const dim3 gpre(kG / 64, kBS / 64);        // big input-projection GEMMs
  const dim3 gstep(kH / 16, (2 * kB) / 64);  // fused recurrent step

  // ---- layer 0: hoisted input projections + fused fwd/bwd recurrence ----
  gemm_bf16<<<gpre, 128, 0, stream>>>(xb,  wT0f, pre_f, kD0P, kD0P, kG);
  gemm_bf16<<<gpre, 128, 0, stream>>>(xrb, wT0b, pre_b, kD0P, kD0P, kG);
  zero_state<<<256, 256, 0, stream>>>(h_st0, c_buf);
  for (int t = 0; t < kS; ++t) {
    unsigned short* hin  = (t & 1) ? h_st1 : h_st0;
    unsigned short* hnew = (t & 1) ? h_st0 : h_st1;
    fused_step<<<gstep, 128, 0, stream>>>(hin, uT0f, uT0b, pre_f, pre_b,
                                          l0f_b, l0b_b, c_buf, hnew,
                                          houtF, houtBr, t);
  }
  combine0<<<(kBS * kD1 + 255) / 256, 256, 0, stream>>>(houtF, houtBr, lens, h0cat, h0rcat);

  // ---- layer 1 ----
  gemm_bf16<<<gpre, 128, 0, stream>>>(h0cat,  wT1f, pre_f, kD1, kD1, kG);
  gemm_bf16<<<gpre, 128, 0, stream>>>(h0rcat, wT1b, pre_b, kD1, kD1, kG);
  zero_state<<<256, 256, 0, stream>>>(h_st0, c_buf);
  for (int t = 0; t < kS; ++t) {
    unsigned short* hin  = (t & 1) ? h_st1 : h_st0;
    unsigned short* hnew = (t & 1) ? h_st0 : h_st1;
    fused_step<<<gstep, 128, 0, stream>>>(hin, uT1f, uT1b, pre_f, pre_b,
                                          l1f_b, l1b_b, c_buf, hnew,
                                          houtF, houtBr, t);
  }
  combine1<<<(kBS * kD1 + 255) / 256, 256, 0, stream>>>(houtF, houtBr, lens, h1cat);

  // ---- FF + emissions + CRF ----
  const dim3 gff(kD1 / 64, kBS / 64);
  gemm_bf16<<<gff, 128, 0, stream>>>(h1cat, ffT, ffpre, kD1, kD1, kD1);
  em_k<<<kBS, kD1, 0, stream>>>(ffpre, ff_b, out_W, out_b, em);
  crf_k<<<1, kB, 0, stream>>>(em, tags, mask, lens, crf_start, crf_end, crf_trans,
                              (float*)d_out);
}
